// NATTENBlock_25357486915720
// MI455X (gfx1250) — matrix operand order repaired and checked by reading, not compiled, not run on hardware
//
#include <hip/hip_runtime.h>
#include <hip/hip_bf16.h>

typedef __attribute__((ext_vector_type(16))) _Float16     v16h;
typedef __attribute__((ext_vector_type(8)))  _Float16     v8h;
typedef __attribute__((ext_vector_type(8)))  float        v8f;
typedef __attribute__((ext_vector_type(4)))  unsigned int v4u;
typedef __attribute__((ext_vector_type(8)))  int          v8i_;
typedef __attribute__((ext_vector_type(4)))  int          v4i_;

// pointer types for the async-to-LDS builtin: AS(1) int4* source, AS(3) int4* dest
typedef __attribute__((address_space(1))) v4i_ gas_v4i;
typedef __attribute__((address_space(3))) v4i_ las_v4i;

#define DIM        128
#define HEADS      8
#define HEAD_DIM   16
#define KS         7
#define GROUPS     8
#define BATCH      4
#define HH         256
#define WW         256
#define NPIX       (HH * WW)                 // 65536
#define M_TOTAL    (BATCH * NPIX)            // 262144
#define QKV_N      (3 * DIM)                 // 384
#define GN_EPS     1e-5f
#define BPAD       136                       // LDS row stride in f16 (272 B -> conflict-free)

#ifndef __has_builtin
#define __has_builtin(x) 0
#endif
#if __has_builtin(__builtin_amdgcn_tensor_load_to_lds) && \
    __has_builtin(__builtin_amdgcn_s_wait_tensorcnt)
#define HAVE_TDM 1
#else
#define HAVE_TDM 0
#endif
#if __has_builtin(__builtin_amdgcn_global_load_async_to_lds_b128)
#define HAVE_ASYNC 1
#else
#define HAVE_ASYNC 0
#endif

// ---------------------------------------------------------------------------
// Stage an (nrows x 128) f16 weight matrix from global into LDS with a padded
// row stride of BPAD f16 (272 B), using the CDNA5 async-to-LDS path when
// available (ASYNCcnt), else plain LDS stores.
// ---------------------------------------------------------------------------
__device__ __forceinline__ void stage_b_lds(_Float16* lds, const _Float16* g, int nrows) {
    const int chunks = nrows * 16;            // 16-byte chunks per row: 256B/row
#if HAVE_ASYNC
    for (int i = threadIdx.x; i < chunks; i += 256) {
        const int row = i >> 4, cb = (i & 15) << 4;   // byte offset within row
        const char* src = (const char*)g + (size_t)row * 256 + cb;
        char*       dst = (char*)lds + (size_t)row * (BPAD * 2) + cb;
        gas_v4i* gp = (gas_v4i*)(unsigned long long)(uintptr_t)src;
        las_v4i* lp = (las_v4i*)(unsigned int)(uintptr_t)dst;
        __builtin_amdgcn_global_load_async_to_lds_b128(gp, lp, 0, 0);
    }
#if __has_builtin(__builtin_amdgcn_s_wait_asynccnt)
    __builtin_amdgcn_s_wait_asynccnt(0);
#else
    asm volatile("s_wait_asynccnt 0x0" ::: "memory");
#endif
#else
    for (int i = threadIdx.x; i < chunks; i += 256) {
        const int row = i >> 4, ce = (i & 15) << 3;   // element offset (8 f16 / chunk)
        *(v8h*)((char*)lds + (size_t)row * (BPAD * 2) + ce * 2) =
            *(const v8h*)(g + (size_t)row * DIM + ce);
    }
#endif
}

#if HAVE_TDM
// ---------------------------------------------------------------------------
// Tensor Data Mover: 1-D copy of nrows*128 f16 (nrows*32 8-byte units) with
// LDS padding: pad_interval=64 dwords (256 B), pad_amount=4 dwords (16 B)
// -> padded LDS row stride 272 B == BPAD f16.  Issued by one wave.
// ---------------------------------------------------------------------------
__device__ __forceinline__ void tdm_load_weights(_Float16* lds, const _Float16* g, int nrows) {
    const unsigned units = (unsigned)nrows * 32u;                 // 8-byte units
    const unsigned lds_addr = (unsigned)(uintptr_t)lds;
    const unsigned long long ga = (unsigned long long)(uintptr_t)g;
    v4u g0;
    g0.x = 1u;                                                    // count=1
    g0.y = lds_addr;                                              // D#.lds_addr
    g0.z = (unsigned)ga;                                          // global_addr lo
    g0.w = (unsigned)(ga >> 32) | (2u << 30);                     // global_addr hi | type=2
    v8i_ g1;
    g1[0] = (int)((3u << 16) | (1u << 20) | (5u << 22) | (3u << 25)); // 8B elems, pad 4dw/64dw
    g1[1] = (int)((units & 0xffffu) << 16);                       // tensor_dim0[15:0]
    g1[2] = (int)(((units >> 16) & 0xffffu) | (1u << 16));        // tensor_dim0 hi | tensor_dim1=1
    g1[3] = (int)((units & 0xffffu) << 16);                       // tile_dim0
    g1[4] = 0;                                                    // tile_dim1/2 unused (1-D)
    g1[5] = (int)units;                                           // tensor_dim0_stride
    g1[6] = 0; g1[7] = 0;
    v4i_ gz = {0, 0, 0, 0};
#if __clang_major__ >= 23
    v8i_ gz8 = {0, 0, 0, 0, 0, 0, 0, 0};
    __builtin_amdgcn_tensor_load_to_lds(g0, g1, gz, gz, gz8, 0);
#else
    __builtin_amdgcn_tensor_load_to_lds(g0, g1, gz, gz, 0);
#endif
    __builtin_amdgcn_s_wait_tensorcnt(0);
}
#endif

// ---------------------------------------------------------------------------
// Kernel 1: per-(batch, group) mean / rstd.  32 blocks, 256 threads.
// ---------------------------------------------------------------------------
__global__ __launch_bounds__(256)
void gn_stats_kernel(const float* __restrict__ x, float* __restrict__ stats) {
    const int bg = blockIdx.x;
    const float* base = x + (size_t)bg * 16 * NPIX;
    const int N  = 16 * NPIX;
    float s = 0.f, s2 = 0.f;
    const float4* b4 = (const float4*)base;
    for (int i = threadIdx.x; i < N / 4; i += 256) {
        float4 v = b4[i];
        s  += v.x + v.y + v.z + v.w;
        s2 += v.x * v.x + v.y * v.y + v.z * v.z + v.w * v.w;
    }
    __shared__ float sh1[256], sh2[256];
    sh1[threadIdx.x] = s; sh2[threadIdx.x] = s2;
    __syncthreads();
    for (int off = 128; off > 0; off >>= 1) {
        if ((int)threadIdx.x < off) {
            sh1[threadIdx.x] += sh1[threadIdx.x + off];
            sh2[threadIdx.x] += sh2[threadIdx.x + off];
        }
        __syncthreads();
    }
    if (threadIdx.x == 0) {
        float mean = sh1[0] / (float)N;
        float var  = sh2[0] / (float)N - mean * mean;
        stats[bg * 2 + 0] = mean;
        stats[bg * 2 + 1] = 1.f / sqrtf(var + GN_EPS);
    }
}

// ---------------------------------------------------------------------------
// Kernel 2: normalize + scale/shift, NCHW f32 -> NHWC f16.
// ---------------------------------------------------------------------------
__global__ __launch_bounds__(256)
void gn_apply_kernel(const float* __restrict__ x, const float* __restrict__ gw,
                     const float* __restrict__ gb, const float* __restrict__ stats,
                     _Float16* __restrict__ xn) {
    const size_t i = (size_t)blockIdx.x * blockDim.x + threadIdx.x;
    const int pix = (int)(i & (NPIX - 1));
    const int c   = (int)((i >> 16) & (DIM - 1));
    const int b   = (int)(i >> 23);
    const int g   = c >> 4;
    const float mean = stats[(b * GROUPS + g) * 2 + 0];
    const float rstd = stats[(b * GROUPS + g) * 2 + 1];
    const float v = (x[i] - mean) * rstd * gw[c] + gb[c];
    xn[((size_t)b * NPIX + pix) * DIM + c] = (_Float16)v;
}

// ---------------------------------------------------------------------------
// Kernel 3: f32 -> f16 weight conversion.
// ---------------------------------------------------------------------------
__global__ __launch_bounds__(256)
void wconv_kernel(const float* __restrict__ wq, const float* __restrict__ wp,
                  _Float16* __restrict__ wq_h, _Float16* __restrict__ wp_h) {
    const int i = blockIdx.x * blockDim.x + threadIdx.x;
    if (i < QKV_N * DIM) wq_h[i] = (_Float16)wq[i];
    const int j = i - QKV_N * DIM;
    if (j >= 0 && j < DIM * DIM) wp_h[j] = (_Float16)wp[j];
}

// ---------------------------------------------------------------------------
// WMMA fragment helpers (16x16x32 f16, wave32).
// A 16x32: lane half selects K sub-octet {0..7,16..23} vs {8..15,24..31}.
// B 32x16: lane half selects K sub-octet {0..7,16..23} vs {8..15,24..31} (same as A).
// C/D f32: lane -> col = lane&15, VGPR j -> row = (lane>>4)*8 + j.
// ---------------------------------------------------------------------------
__device__ __forceinline__ v16h load_a_frag(const _Float16* __restrict__ rowp) {
    v8h lo = *(const v8h*)(rowp);
    v8h hi = *(const v8h*)(rowp + 16);
    return __builtin_shufflevector(lo, hi, 0,1,2,3,4,5,6,7,8,9,10,11,12,13,14,15);
}

__device__ __forceinline__ v16h load_b_frag_lds(const _Float16* bp) {
    v8h lo = *(const v8h*)(bp);
    v8h hi = *(const v8h*)(bp + 16);
    return __builtin_shufflevector(lo, hi, 0,1,2,3,4,5,6,7,8,9,10,11,12,13,14,15);
}

// ---------------------------------------------------------------------------
// Kernel 4: QKV GEMM.  out[m][n] = sum_k xn[m][k] * qkv_w[n][k].
// Weights staged into LDS once per block (TDM or async-to-LDS), then each of
// the 8 waves computes one 16-row M-tile across all 24 N-tiles.
// ---------------------------------------------------------------------------
__global__ __launch_bounds__(256)
void qkv_gemm_kernel(const _Float16* __restrict__ xn,
                     const _Float16* __restrict__ wq_h,
                     _Float16* __restrict__ qkv) {
    extern __shared__ _Float16 bsh[];            // QKV_N * BPAD f16 = 104448 B
#if HAVE_TDM
    if (threadIdx.x < 32) tdm_load_weights(bsh, wq_h, QKV_N);
#else
    stage_b_lds(bsh, wq_h, QKV_N);
#endif
    __syncthreads();

    const int lane = threadIdx.x & 31;
    const int wid  = blockIdx.x * (blockDim.x >> 5) + (threadIdx.x >> 5);
    const size_t m0 = (size_t)wid * 16;
    const int r    = lane & 15;
    const int kb   = (lane >> 4) << 3;
    const int cc   = lane & 15;
    const int kbb  = (lane >> 4) << 3;
    const int half = (lane >> 4) << 3;

    v16h a[4];
    const _Float16* ap = xn + (m0 + r) * DIM + kb;
#pragma unroll
    for (int ks = 0; ks < 4; ++ks) a[ks] = load_a_frag(ap + ks * 32);

    for (int nt = 0; nt < QKV_N / 16; ++nt) {
        v8f acc = {};
        const _Float16* bp = bsh + (size_t)(nt * 16 + cc) * BPAD + kbb;
#pragma unroll
        for (int ks = 0; ks < 4; ++ks) {
            v16h b = load_b_frag_lds(bp + ks * 32);
            acc = __builtin_amdgcn_wmma_f32_16x16x32_f16(
                false, a[ks], false, b, (short)0, acc, false, false);
        }
        _Float16* op = qkv + (m0 + half) * QKV_N + nt * 16 + cc;
#pragma unroll
        for (int j = 0; j < 8; ++j) op[(size_t)j * QKV_N] = (_Float16)acc[j];
    }
}

// ---------------------------------------------------------------------------
// Kernel 5: 7x7 neighborhood attention.  One lane per (pixel, head).
// ---------------------------------------------------------------------------
__global__ __launch_bounds__(256)
void natten_kernel(const _Float16* __restrict__ qkv, _Float16* __restrict__ o) {
    const size_t t = (size_t)blockIdx.x * blockDim.x + threadIdx.x;
    const int head = (int)(t & (HEADS - 1));
    const size_t p = t >> 3;
    const int xx = (int)(p & (WW - 1));
    const int yy = (int)((p >> 8) & (HH - 1));
    const int b  = (int)(p >> 16);

    float q[HEAD_DIM];
    {
        v16h qv = *(const v16h*)(qkv + p * QKV_N + head * HEAD_DIM);
#pragma unroll
        for (int i = 0; i < HEAD_DIM; ++i) q[i] = (float)qv[i];
    }
    int sh = yy - KS / 2; sh = sh < 0 ? 0 : (sh > HH - KS ? HH - KS : sh);
    int sw = xx - KS / 2; sw = sw < 0 ? 0 : (sw > WW - KS ? WW - KS : sw);

    float logits[KS * KS];
    float mx = -3.4e38f;
#pragma unroll
    for (int di = 0; di < KS; ++di) {
        const size_t rowbase =
            (((size_t)b * HH + (sh + di)) * WW + sw) * QKV_N + DIM + head * HEAD_DIM;
#pragma unroll
        for (int dj = 0; dj < KS; ++dj) {
            v16h kv = *(const v16h*)(qkv + rowbase + (size_t)dj * QKV_N);
            float dot = 0.f;
#pragma unroll
            for (int i = 0; i < HEAD_DIM; ++i) dot += q[i] * (float)kv[i];
            dot *= 0.25f;
            logits[di * KS + dj] = dot;
            mx = fmaxf(mx, dot);
        }
    }
    float s = 0.f;
#pragma unroll
    for (int i = 0; i < KS * KS; ++i) {
        float e = __expf(logits[i] - mx);
        logits[i] = e;
        s += e;
    }
    const float inv = 1.f / s;

    float acc[HEAD_DIM];
#pragma unroll
    for (int i = 0; i < HEAD_DIM; ++i) acc[i] = 0.f;
#pragma unroll
    for (int di = 0; di < KS; ++di) {
        const size_t rowbase =
            (((size_t)b * HH + (sh + di)) * WW + sw) * QKV_N + 2 * DIM + head * HEAD_DIM;
#pragma unroll
        for (int dj = 0; dj < KS; ++dj) {
            v16h vv = *(const v16h*)(qkv + rowbase + (size_t)dj * QKV_N);
            const float w = logits[di * KS + dj] * inv;
#pragma unroll
            for (int i = 0; i < HEAD_DIM; ++i) acc[i] += w * (float)vv[i];
        }
    }
    v16h ov;
#pragma unroll
    for (int i = 0; i < HEAD_DIM; ++i) ov[i] = (_Float16)acc[i];
    *(v16h*)(o + p * DIM + head * HEAD_DIM) = ov;
}

// ---------------------------------------------------------------------------
// Kernel 6: proj GEMM + residual.  out = x + (o @ proj_w.T) * gamma, NCHW f32.
// Weights (128x128) staged into static LDS via async-to-LDS.
// ---------------------------------------------------------------------------
__global__ __launch_bounds__(256)
void proj_gemm_kernel(const _Float16* __restrict__ o_h,
                      const _Float16* __restrict__ wp_h,
                      const float* __restrict__ x,
                      const float* __restrict__ gamma,
                      float* __restrict__ out) {
    __shared__ _Float16 bsh[DIM * BPAD];         // 34816 B
    stage_b_lds(bsh, wp_h, DIM);
    __syncthreads();

    const int lane = threadIdx.x & 31;
    const int wid  = blockIdx.x * (blockDim.x >> 5) + (threadIdx.x >> 5);
    const size_t m0 = (size_t)wid * 16;
    const int r    = lane & 15;
    const int kb   = (lane >> 4) << 3;
    const int cc   = lane & 15;
    const int kbb  = (lane >> 4) << 3;
    const int half = (lane >> 4) << 3;
    const float g  = gamma[0];

    v16h a[4];
    const _Float16* ap = o_h + (m0 + r) * DIM + kb;
#pragma unroll
    for (int ks = 0; ks < 4; ++ks) a[ks] = load_a_frag(ap + ks * 32);

    for (int nt = 0; nt < DIM / 16; ++nt) {
        v8f acc = {};
        const _Float16* bp = bsh + (size_t)(nt * 16 + cc) * BPAD + kbb;
#pragma unroll
        for (int ks = 0; ks < 4; ++ks) {
            v16h b = load_b_frag_lds(bp + ks * 32);
            acc = __builtin_amdgcn_wmma_f32_16x16x32_f16(
                false, a[ks], false, b, (short)0, acc, false, false);
        }
        const int c = nt * 16 + cc;
#pragma unroll
        for (int j = 0; j < 8; ++j) {
            const size_t m  = m0 + half + j;
            const int bb    = (int)(m >> 16);
            const int pix   = (int)(m & (NPIX - 1));
            const size_t ix = ((size_t)bb * DIM + c) * NPIX + pix;
            out[ix] = x[ix] + acc[j] * g;
        }
    }
}

// ---------------------------------------------------------------------------
// Host launcher.  Workspace: stats(256B) | wq f16(96KB) | wp f16(32KB) |
// xn/o f16 (64MB, aliased) | qkv f16 (192MB).  ~269 MB total.
// ---------------------------------------------------------------------------
extern "C" void kernel_launch(void* const* d_in, const int* in_sizes, int n_in,
                              void* d_out, int out_size, void* d_ws, size_t ws_size,
                              hipStream_t stream) {
    const float* x      = (const float*)d_in[0];
    const float* gn_w   = (const float*)d_in[1];
    const float* gn_b   = (const float*)d_in[2];
    const float* qkv_w  = (const float*)d_in[3];
    const float* proj_w = (const float*)d_in[4];
    const float* gamma  = (const float*)d_in[5];
    float* out          = (float*)d_out;

    char* ws = (char*)d_ws;
    float*    stats = (float*)ws;
    _Float16* wq_h  = (_Float16*)(ws + 256);
    _Float16* wp_h  = (_Float16*)(ws + 256 + 98304);
    _Float16* xn_h  = (_Float16*)(ws + 131328);                 // reused as o_h
    _Float16* qkv_h = (_Float16*)(ws + 131328 + 67108864ull);

    gn_stats_kernel<<<BATCH * GROUPS, 256, 0, stream>>>(x, stats);

    gn_apply_kernel<<<(BATCH * DIM * NPIX) / 256, 256, 0, stream>>>(
        x, gn_w, gn_b, stats, xn_h);

    wconv_kernel<<<(QKV_N * DIM + DIM * DIM) / 256, 256, 0, stream>>>(
        qkv_w, proj_w, wq_h, wp_h);

    qkv_gemm_kernel<<<(M_TOTAL / 16) / 8, 256, (size_t)(QKV_N * BPAD * 2), stream>>>(
        xn_h, wq_h, qkv_h);

    natten_kernel<<<(M_TOTAL * HEADS) / 256, 256, 0, stream>>>(qkv_h, xn_h);

    proj_gemm_kernel<<<(M_TOTAL / 16) / 8, 256, 0, stream>>>(
        xn_h, wp_h, x, gamma, out);
}
